// SSformer_73650099191990
// MI455X (gfx1250) — matrix operand order repaired and checked by reading, not compiled, not run on hardware
//
#include <hip/hip_runtime.h>
#include <hip/hip_bf16.h>

// MI455X / gfx1250, wave32. All GEMMs via v_wmma_f32_16x16x32_f16.
// Each wave computes a 16x64 output tile (1 A fragment reused over 4 B fragments).

typedef __attribute__((ext_vector_type(16))) _Float16 v16h;
typedef __attribute__((ext_vector_type(8)))  _Float16 v8h;
typedef __attribute__((ext_vector_type(8)))  float    v8f;

#define IDX1D (long)blockIdx.x * blockDim.x + threadIdx.x

// ---------------- elementwise / layout kernels ----------------

__global__ void k_cvt(const float* __restrict__ x, _Float16* __restrict__ y, long n) {
  long i = IDX1D; if (i < n) y[i] = (_Float16)x[i];
}

// wt[n*K + k] = w[k*N + n]  (fp32 KxN -> f16 NxK, pre-transposed B operand)
__global__ void k_wT(const float* __restrict__ w, _Float16* __restrict__ wt, int K, int N, long n) {
  long i = IDX1D; if (i >= n) return;
  int nn = (int)(i % N); long kk = i / N;
  wt[(long)nn * K + kk] = (_Float16)w[i];
}

// (B, 512, 16,16) NCHW -> (B*256, 512) rows, f16
__global__ void k_permcvt(const float* __restrict__ in, _Float16* __restrict__ out, long n) {
  long i = IDX1D; if (i >= n) return;
  int c = (int)(i & 511); int t = (int)((i >> 9) & 255); int b = (int)(i >> 17);
  out[i] = (_Float16) in[((long)b * 512 + c) * 256 + t];
}

// (B*256, 512) fp32 -> (B, 512, 256) fp32 output
__global__ void k_permout(const float* __restrict__ d, float* __restrict__ out, long n) {
  long i = IDX1D; if (i >= n) return;
  int t = (int)(i & 255); int c = (int)((i >> 8) & 511); int b = (int)(i >> 17);
  out[i] = d[((long)b * 256 + t) * 512 + c];
}

// Per-(b,h) transposed V, f16: vt[(b*8+h)*64*256 + d*256 + j] = v[b,jj,h*64+d]
// perm=1 applies the fourier-collapse permutation jj = (256 - j) & 255.
__global__ void k_vT(const float* __restrict__ src, _Float16* __restrict__ vt,
                     int ldsrc, int colOff, int perm, long n) {
  long i = IDX1D; if (i >= n) return;
  int j = (int)(i & 255); int dd = (int)((i >> 8) & 63); int z = (int)(i >> 14);
  int h = z & 7; int b = z >> 3;
  int jj = perm ? ((256 - j) & 255) : j;
  vt[i] = (_Float16) src[((long)(b * 256 + jj)) * ldsrc + colOff + h * 64 + dd];
}

// -------- LayerNorm: one wave per 512-wide row, f16 output (feeds WMMA A) --------
__global__ void k_ln(const float* __restrict__ x, _Float16* __restrict__ y,
                     const float* __restrict__ g, const float* __restrict__ b, int rows) {
  int row = blockIdx.x * blockDim.y + threadIdx.y;
  if (row >= rows) return;
  int lane = threadIdx.x;
  const float* xr = x + (long)row * 512;
  float v[16]; float s = 0.f;
#pragma unroll
  for (int i = 0; i < 16; i++) { v[i] = xr[lane + i * 32]; s += v[i]; }
#pragma unroll
  for (int o = 16; o > 0; o >>= 1) s += __shfl_xor(s, o, 32);
  float mean = s * (1.f / 512.f);
  float vs = 0.f;
#pragma unroll
  for (int i = 0; i < 16; i++) { float d = v[i] - mean; vs += d * d; }
#pragma unroll
  for (int o = 16; o > 0; o >>= 1) vs += __shfl_xor(vs, o, 32);
  float inv = rsqrtf(vs * (1.f / 512.f) + 1e-5f);
  _Float16* yr = y + (long)row * 512;
#pragma unroll
  for (int i = 0; i < 16; i++) {
    int c = lane + i * 32;
    yr[c] = (_Float16)((v[i] - mean) * inv * g[c] + b[c]);
  }
}

// -------- softmax over 256-wide rows, f16 output -----------------
// p = softmax( (doAbs ? |x| : x) * inScale ) * outScale
__global__ void k_softmax(const float* __restrict__ x, _Float16* __restrict__ p,
                          int rows, float inScale, int doAbs, float outScale) {
  int row = blockIdx.x * blockDim.y + threadIdx.y;
  if (row >= rows) return;
  int lane = threadIdx.x;
  const float* xr = x + (long)row * 256;
  float v[8]; float mx = -3.4e38f;
#pragma unroll
  for (int i = 0; i < 8; i++) {
    float t = xr[lane + i * 32];
    t = doAbs ? fabsf(t) * inScale : t * inScale;
    v[i] = t; mx = fmaxf(mx, t);
  }
#pragma unroll
  for (int o = 16; o > 0; o >>= 1) mx = fmaxf(mx, __shfl_xor(mx, o, 32));
  float s = 0.f;
#pragma unroll
  for (int i = 0; i < 8; i++) { v[i] = __expf(v[i] - mx); s += v[i]; }
#pragma unroll
  for (int o = 16; o > 0; o >>= 1) s += __shfl_xor(s, o, 32);
  float r = outScale / s;
  _Float16* pr = p + (long)row * 256;
#pragma unroll
  for (int i = 0; i < 8; i++) pr[lane + i * 32] = (_Float16)(v[i] * r);
}

// --- per-sample standardize (unbiased var, no eps) + AdaLN, f16 output ----
__global__ void k_stdadaln(const float* __restrict__ x, const float* __restrict__ t,
                           _Float16* __restrict__ y) {
  int b = blockIdx.x; int tid = threadIdx.x;
  const float* xb = x + (long)b * (256 * 512);
  float s = 0.f, ss = 0.f;
  for (int i = tid; i < 256 * 512; i += 256) { float v = xb[i]; s += v; ss += v * v; }
  __shared__ float sh1[256], sh2[256];
  sh1[tid] = s; sh2[tid] = ss; __syncthreads();
  for (int o = 128; o > 0; o >>= 1) {
    if (tid < o) { sh1[tid] += sh1[tid + o]; sh2[tid] += sh2[tid + o]; }
    __syncthreads();
  }
  float S = sh1[0], SS = sh2[0];
  const float n = 131072.f;
  float mean = S / n;
  float var = (SS - S * mean) / (n - 1.f);
  float inv = rsqrtf(var);
  const float* m = t + (long)b * 1024;
  const float* sc = m + 512;
  _Float16* yb = y + (long)b * (256 * 512);
  for (int i = tid; i < 256 * 512; i += 256) {
    int c = i & 511;
    yb[i] = (_Float16)((xb[i] - mean) * inv * sc[c] + m[c]);
  }
}

// ------------- batched WMMA GEMM: C = A(MxK) * Bt(NxK)^T --------------------
// 4 waves/block; each wave owns a 16(M)x64(N) tile: A fragment loaded once per
// k-step, reused over 4 B fragments / 4 accumulators.
// Composite batch offsets: off = (z/innerDiv)*outer + (z%innerDiv)*inner.
// C (fp32) and/or C16 (f16) outputs, either may be null.
// act: 0 none, 1 exact GELU, 2 SiLU. pos adds pos[(row%256)*N+col].
__global__ __launch_bounds__(128) void k_gemm(
    const _Float16* __restrict__ A, const _Float16* __restrict__ Bt,
    float* __restrict__ C, _Float16* __restrict__ C16,
    const float* __restrict__ bias, const float* __restrict__ pos,
    int M, int N, int K, int lda, int ldb, int ldc,
    int innerDiv, long aO, long aI, long bO, long bI, long cO, long cI,
    int act)
{
  int lane = threadIdx.x;
  int tm = blockIdx.y * 4 + threadIdx.y;     // wave-uniform guard: EXEC stays all-ones
  if (tm * 16 >= M) return;
  int tn4 = blockIdx.x;                      // 64-wide column group
  int z = blockIdx.z;
  int zo = z / innerDiv, zi = z - zo * innerDiv;
  long aOff = (long)zo * aO + (long)zi * aI;
  long bOff = (long)zo * bO + (long)zi * bI;
  long cOff = (long)zo * cO + (long)zi * cI;
  int hi = lane >> 4;
  int l  = lane & 15;
  // A fragment: row = l, K halves {0..7,16..23} (lo lanes) / {8..15,24..31} (hi lanes)
  const _Float16* aP = A + aOff + (long)(tm * 16 + l) * lda + hi * 8;
  // B fragments: col = l, K halves {0..7,16..23} (lo lanes) / {8..15,24..31} (hi lanes)
  const _Float16* bP0 = Bt + bOff + (long)(tn4 * 64 + l) * ldb + hi * 8;
  v8f acc[4] = {};
  for (int k0 = 0; k0 < K; k0 += 32) {
    v8h a0 = *(const v8h*)(aP + k0);
    v8h a1 = *(const v8h*)(aP + k0 + 16);
    v16h av = __builtin_shufflevector(a0, a1, 0,1,2,3,4,5,6,7,8,9,10,11,12,13,14,15);
#pragma unroll
    for (int j = 0; j < 4; j++) {
      const _Float16* bP = bP0 + (long)(j * 16) * ldb;
      v8h b0 = *(const v8h*)(bP + k0);
      v8h b1 = *(const v8h*)(bP + k0 + 16);
      v16h bv = __builtin_shufflevector(b0, b1, 0,1,2,3,4,5,6,7,8,9,10,11,12,13,14,15);
      acc[j] = __builtin_amdgcn_wmma_f32_16x16x32_f16(false, av, false, bv,
                                                      (short)0, acc[j], false, false);
    }
  }
  int rb = tm * 16 + hi * 8;
#pragma unroll
  for (int j = 0; j < 4; j++) {
    int col = tn4 * 64 + j * 16 + l;
    float badd = bias ? bias[col] : 0.f;
#pragma unroll
    for (int r = 0; r < 8; r++) {
      int row = rb + r;
      float v = acc[j][r] + badd;
      if (pos) v += pos[(row & 255) * N + col];
      if (act == 1)      v = 0.5f * v * (1.f + erff(v * 0.70710678118654752f));
      else if (act == 2) v = v / (1.f + __expf(-v));
      long idx = cOff + (long)row * ldc + col;
      if (C)   C[idx]   = v;
      if (C16) C16[idx] = (_Float16)v;
    }
  }
}

// =======================================================================

static inline void launch_gemm(hipStream_t st,
    const _Float16* A, int lda, const _Float16* Bt, int ldb,
    float* C, _Float16* C16, int ldc, int M, int N, int K,
    const float* bias, const float* pos, int act,
    int batch = 1, int innerDiv = 1,
    long aO = 0, long aI = 0, long bO = 0, long bI = 0, long cO = 0, long cI = 0)
{
  dim3 blk(32, 4, 1), grd(N / 64, (M + 63) / 64, batch);
  k_gemm<<<grd, blk, 0, st>>>(A, Bt, C, C16, bias, pos, M, N, K, lda, ldb, ldc,
                              innerDiv, aO, aI, bO, bI, cO, cI, act);
}

extern "C" void kernel_launch(void* const* d_in, const int* in_sizes, int n_in,
                              void* d_out, int out_size, void* d_ws, size_t ws_size,
                              hipStream_t stream) {
  (void)in_sizes; (void)n_in; (void)out_size; (void)ws_size;
  const int B = 16, H = 8;
  const long M = 4096;             // B * 256 token-rows
  const int D = 512, D3 = 1536;

  const float* con  = (const float*)d_in[0];
  const float* diff = (const float*)d_in[1];
  const float* temb = (const float*)d_in[2];
  auto P = [&](int i) { return (const float*)d_in[i]; };
  // flatten map (recursive dict-insertion order):
  // 3 c_lproj_w, 4 c_lproj_b, 5 d_lproj_w, 6 d_lproj_b, 7 c_pos, 8 d_pos
  // per block (base 9+30*blk):
  //  +0 norm_g +1 norm_b +2 qkv_w +3 out_w +4 out_b
  //  +5 ff.ln_g +6 ff.ln_b +7 ff.w1 +8 ff.b1 +9 ff.w2 +10 ff.b2
  //  +11 ln_diff_g +12 ln_diff_b +13 ln_con_g +14 ln_con_b
  //  +15 q_w +16 k_w +17 v_w +18 out_w +19 out_b
  //  +20 emd1_w +21 emd1_b +22 emd2_w +23 emd2_b
  //  +24 mlp.ln_g +25 mlp.ln_b +26 mlp.w1 +27 mlp.b1 +28 mlp.w2 +29 mlp.b2

  // ---- workspace allocator ----
  char* base = (char*)d_ws;
  size_t cur = 0;
  auto alloc = [&](size_t bytes) -> void* {
    void* p = base + cur;
    cur = (cur + bytes + 255) & ~(size_t)255;
    return p;
  };
  float* c32 = (float*)alloc(M * D * 4);
  float* d32 = (float*)alloc(M * D * 4);
  float* x1  = (float*)alloc(M * D3 * 4);            // qkv fp32 (for V extraction)
  float* x2  = (float*)alloc(M * D * 4);
  float* x3  = (float*)alloc(M * D * 4);
  float* s32 = (float*)alloc((long)B * H * 256 * 256 * 4);
  float* t32b = (float*)alloc((long)B * 1024 * 4);
  _Float16* af16 = (_Float16*)alloc((long)B * H * 256 * 256 * 2);  // LN out / probs
  _Float16* bf16 = (_Float16*)alloc(M * D3 * 2);                   // qkv f16 / 3 MxD slots
  _Float16* vt16 = (_Float16*)alloc((long)B * H * 64 * 256 * 2);
  _Float16* tE16 = (_Float16*)alloc((long)B * 512 * 2);
  _Float16* t16  = (_Float16*)alloc((long)B * 1024 * 2);

  auto wT = [&](const float* w, int K, int N) -> _Float16* {
    _Float16* dst = (_Float16*)alloc((size_t)K * N * 2);
    long tot = (long)K * N;
    k_wT<<<(tot + 255) / 256, 256, 0, stream>>>(w, dst, K, N, tot);
    return dst;
  };
  _Float16* Wcl = wT(P(3), D, D);
  _Float16* Wdl = wT(P(5), D, D);
  struct BW { _Float16 *qkv, *aout, *fw1, *fw2, *q, *k, *v, *cout, *e1, *e2, *m1, *m2; } bw[4];
  for (int blk = 0; blk < 4; blk++) {
    int pi = 9 + 30 * blk;
    bw[blk].qkv  = wT(P(pi + 2),  D, D3);
    bw[blk].aout = wT(P(pi + 3),  D, D);
    bw[blk].fw1  = wT(P(pi + 7),  D, D);
    bw[blk].fw2  = wT(P(pi + 9),  D, D);
    bw[blk].q    = wT(P(pi + 15), D, D);
    bw[blk].k    = wT(P(pi + 16), D, D);
    bw[blk].v    = wT(P(pi + 17), D, D);
    bw[blk].cout = wT(P(pi + 18), D, D);
    bw[blk].e1   = wT(P(pi + 20), 512, 1024);
    bw[blk].e2   = wT(P(pi + 22), 1024, 1024);
    bw[blk].m1   = wT(P(pi + 26), D, D);
    bw[blk].m2   = wT(P(pi + 28), D, D);
  }

  auto ln = [&](const float* x, _Float16* y, const float* g, const float* b) {
    k_ln<<<4096 / 8, dim3(32, 8), 0, stream>>>(x, y, g, b, 4096);
  };
  auto softmax = [&](const float* x, _Float16* p, float inS, int doAbs, float outS) {
    k_softmax<<<32768 / 8, dim3(32, 8), 0, stream>>>(x, p, 32768, inS, doAbs, outS);
  };

  long nMD = M * D, nVT = (long)B * H * 64 * 256;
  _Float16* s0 = bf16;            // 3 MxD f16 slots
  _Float16* s1 = bf16 + nMD;
  _Float16* s2 = bf16 + 2 * nMD;

  // ---- input projections: x @ W + b + pos ----
  k_permcvt<<<(nMD + 255) / 256, 256, 0, stream>>>(con, af16, nMD);
  launch_gemm(stream, af16, D, Wcl, D, c32, nullptr, D, 4096, D, D, P(4), P(7), 0);
  k_permcvt<<<(nMD + 255) / 256, 256, 0, stream>>>(diff, af16, nMD);
  launch_gemm(stream, af16, D, Wdl, D, d32, nullptr, D, 4096, D, D, P(6), P(8), 0);
  k_cvt<<<(B * 512 + 255) / 256, 256, 0, stream>>>(temb, tE16, (long)B * 512);

  for (int blk = 0; blk < 4; blk++) {
    int pi = 9 + 30 * blk;
    // ===== self-attention on c (pre-LN, no residual) =====
    ln(c32, af16, P(pi + 0), P(pi + 1));
    launch_gemm(stream, af16, D, bw[blk].qkv, D, x1, bf16, D3, 4096, D3, D, nullptr, nullptr, 0);
    // scores[b,h] = q @ k^T  (q cols [0,512), k cols [512,1024) inside qkv)
    launch_gemm(stream, bf16, D3, bf16 + 512, D3, s32, nullptr, 256, 256, 256, 64,
                nullptr, nullptr, 0, B * H, H,
                256L * D3, 64, 256L * D3, 64, (long)H * 65536, 65536);
    softmax(s32, af16, 0.125f, 0, 1.0f);
    k_vT<<<(nVT + 255) / 256, 256, 0, stream>>>(x1, vt16, D3, 1024, 0, nVT);
    // merged[b, n, h*64+d] = probs @ v  -> f16 directly (feeds out-proj)
    launch_gemm(stream, af16, 256, vt16, 256, nullptr, s0, D, 256, 64, 256,
                nullptr, nullptr, 0, B * H, H,
                (long)H * 65536, 65536, (long)H * 16384, 16384, 256L * D, 64);
    launch_gemm(stream, s0, D, bw[blk].aout, D, x3, nullptr, D, 4096, D, D, P(pi + 4), nullptr, 0);
    // FeedForward
    ln(x3, s0, P(pi + 5), P(pi + 6));
    launch_gemm(stream, s0, D, bw[blk].fw1, D, nullptr, s1, D, 4096, D, D, P(pi + 8), nullptr, 1);
    launch_gemm(stream, s1, D, bw[blk].fw2, D, c32, nullptr, D, 4096, D, D, P(pi + 10), nullptr, 0);

    // ===== cross "fourier" attention (collapses to abs-softmax attn + perm V) =====
    // time MLP
    launch_gemm(stream, tE16, 512, bw[blk].e1, 512, nullptr, t16, 1024, 16, 1024, 512,
                P(pi + 21), nullptr, 2);
    launch_gemm(stream, t16, 1024, bw[blk].e2, 1024, t32b, nullptr, 1024, 16, 1024, 1024,
                P(pi + 23), nullptr, 0);

    ln(d32, s0, P(pi + 11), P(pi + 12));           // fq  = LN(diff) -> slot0
    ln(c32, af16, P(pi + 13), P(pi + 14));         // fkv = LN(con)  -> af16
    launch_gemm(stream, af16, D, bw[blk].k, D, nullptr, s1, D, 4096, D, D, nullptr, nullptr, 0);
    launch_gemm(stream, af16, D, bw[blk].v, D, x3, nullptr, D, 4096, D, D, nullptr, nullptr, 0);
    k_vT<<<(nVT + 255) / 256, 256, 0, stream>>>(x3, vt16, D, 0, 1, nVT);  // permuted V^T
    launch_gemm(stream, s0, D, bw[blk].q, D, nullptr, s2, D, 4096, D, D, nullptr, nullptr, 0);
    launch_gemm(stream, s2, D, s1, D, s32, nullptr, 256, 256, 256, 64,
                nullptr, nullptr, 0, B * H, H,
                256L * D, 64, 256L * D, 64, (long)H * 65536, 65536);
    softmax(s32, af16, 1.0f, 1, 0.125f);           // softmax(|s|) / sqrt(64)
    launch_gemm(stream, af16, 256, vt16, 256, x2, nullptr, D, 256, 64, 256,
                nullptr, nullptr, 0, B * H, H,
                (long)H * 65536, 65536, (long)H * 16384, 16384, 256L * D, 64);
    k_stdadaln<<<B, 256, 0, stream>>>(x2, t32b, s0);
    launch_gemm(stream, s0, D, bw[blk].cout, D, x2, nullptr, D, 4096, D, D, P(pi + 19), nullptr, 0);
    // mlp FeedForward
    ln(x2, s0, P(pi + 24), P(pi + 25));
    launch_gemm(stream, s0, D, bw[blk].m1, D, nullptr, s1, D, 4096, D, D, P(pi + 27), nullptr, 1);
    launch_gemm(stream, s1, D, bw[blk].m2, D, d32, nullptr, D, 4096, D, D, P(pi + 29), nullptr, 0);
  }

  long nOut = (long)B * 512 * 256;
  k_permout<<<(nOut + 255) / 256, 256, 0, stream>>>(d32, (float*)d_out, nOut);
}
